// Generator_cor_map_84413287236131
// MI455X (gfx1250) — matrix operand order repaired and checked by reading, not compiled, not run on hardware
//
#include <hip/hip_runtime.h>
#include <cstdint>
#include <cstddef>

#define LEN_SEQ 20
#define NTOK    4096
#define BATCH   16
#define HDN     100
#define EMBD    100
#define INV_BN  0.9999950000374997f   // 1/sqrt(1+1e-5)
#define KCHUNK  1024                  // A-panel K-chunk in LDS (16*1024*2 = 32 KB/panel)
#define KPAD2   640                   // log_W2 K padded 600 -> 640 (multiple of 128)

typedef __bf16 bf16x16 __attribute__((ext_vector_type(16)));
typedef float  floatx8 __attribute__((ext_vector_type(8)));

union FragAB { bf16x16 v; uint4 q[2]; };

static __device__ __forceinline__ unsigned short f2bf(float f) {
  unsigned int u = __float_as_uint(f);
  u += 0x7FFFu + ((u >> 16) & 1u);        // round-to-nearest-even
  return (unsigned short)(u >> 16);
}
static __device__ __forceinline__ float sigmoidf_(float x) { return 1.f / (1.f + __expf(-x)); }

// ---------------------------------------------------------------------------
// Convert fp32 weight matrix [rows, scol] -> bf16 [rows, dcol] (zero-padded K)
// ---------------------------------------------------------------------------
__global__ __launch_bounds__(256) void k_convert_pad(
    const float* __restrict__ src, unsigned short* __restrict__ dst,
    int rows, int scol, int dcol) {
  long long i = (long long)blockIdx.x * 256 + threadIdx.x;
  long long total = (long long)rows * dcol;
  if (i >= total) return;
  int r = (int)(i / dcol), cc = (int)(i % dcol);
  float v = (cc < scol) ? src[(long long)r * scol + cc] : 0.f;
  dst[i] = f2bf(v);
}

// ---------------------------------------------------------------------------
// WMMA bf16 GEMM: C[16,N] = act(A[16,K] * W[N,K]^T + bias)
//   - A panel double-buffered in LDS via async global->LDS copies (ASYNCcnt):
//     panel p+1 staged while panel p is consumed (ds_load_b128)
//   - B streamed from global/L2 with clustered b128 loads + prefetch
//   - K swept 128/iteration: 4 fragment pairs loaded, then 4 back-to-back
//     v_wmma_f32_16x16x32_bf16 (software pipelined)
//   act==0: ELU -> bf16 out; act==1: sigmoid -> f32 out
//   Requires K % 128 == 0 (and KCHUNK % 128 == 0).
// ---------------------------------------------------------------------------
__global__ __launch_bounds__(256) void k_wmma_gemm(
    const unsigned short* __restrict__ Abf,   // [16, K] bf16 row-major
    const unsigned short* __restrict__ Wbf,   // [N,  K] bf16 row-major
    const float* __restrict__ bias,           // [N]
    void* __restrict__ out, int K, int N, int act) {
  __shared__ unsigned short sA[2 * 16 * KCHUNK];  // 2 x 32 KB A panels

  const int tid  = threadIdx.x;
  const int wave = (int)((blockIdx.x * blockDim.x + tid) >> 5);
  const int lane = tid & 31;
  const int half = lane >> 4;
  const int lm   = lane & 15;
  int ncol0 = wave * 16;
  if (ncol0 > N - 16) ncol0 = N - 16;        // keep loads in-bounds; no early return
                                             // (kernel has barriers; grids are exact)
  const unsigned short* brow = Wbf + (size_t)(ncol0 + lm) * K;
  floatx8 acc = {0.f, 0.f, 0.f, 0.f, 0.f, 0.f, 0.f, 0.f};
  const uint32_t ldsBase = (uint32_t)(uintptr_t)(const void*)sA;

  // async-stage A[16, p*KCHUNK .. ) into LDS panel (p&1); full-EXEC, no divides
  const int srow = tid >> 4;                 // 0..15
  const int sc0  = tid & 15;
  auto stageA = [&](int p) {
    const int kc = p * KCHUNK;
    int clen = K - kc; if (clen > KCHUNK) clen = KCHUNK;
    const int c8n = clen >> 3;               // b128 packets per row
    const unsigned short* gp = Abf + (size_t)srow * K + kc;
    const uint32_t lbase = ldsBase +
        (uint32_t)(((p & 1) * 16 * KCHUNK + srow * KCHUNK) * 2);
    for (int c8 = sc0; c8 < c8n; c8 += 16) {
      uint64_t ga = (uint64_t)(uintptr_t)(gp + (c8 << 3));
      uint32_t la = lbase + (uint32_t)(c8 << 4);
      asm volatile("global_load_async_to_lds_b128 %0, %1, off"
                   :: "v"(la), "v"(ga) : "memory");
    }
  };

  const int P = (K + KCHUNK - 1) / KCHUNK;
  stageA(0);
  for (int p = 0; p < P; ++p) {
    asm volatile("s_wait_asynccnt 0x0" ::: "memory");  // panel p staged
    __syncthreads();                                    // visible; prev panel consumed
    if (p + 1 < P) stageA(p + 1);                       // overlap copy with compute
    const int kc = p * KCHUNK;
    int clen = K - kc; if (clen > KCHUNK) clen = KCHUNK;
    const unsigned short* base = sA + (p & 1) * 16 * KCHUNK + lm * KCHUNK;
    for (int k0 = 0; k0 < clen; k0 += 128) {
      FragAB fa[4], fb[4];
#pragma unroll
      for (int uu = 0; uu < 4; ++uu) {
        // A 16x32 layout: lane half h holds 16 contiguous K at 16h
        const unsigned short* la = base + k0 + uu * 32;
        fa[uu].q[0] = *(const uint4*)(la + half * 16);
        fa[uu].q[1] = *(const uint4*)(la + half * 16 + 8);
        // B 32x16 layout: lane half h holds 16 contiguous K at 16h
        const unsigned short* lb = brow + kc + k0 + uu * 32;
        fb[uu].q[0] = *(const uint4*)(lb + half * 16);
        fb[uu].q[1] = *(const uint4*)(lb + half * 16 + 8);
      }
      if (kc + k0 + 256 < K) __builtin_prefetch(brow + kc + k0 + 256, 0, 3);
#pragma unroll
      for (int uu = 0; uu < 4; ++uu)
        acc = __builtin_amdgcn_wmma_f32_16x16x32_bf16(
            /*neg_a=*/false, fa[uu].v, /*neg_b=*/false, fb[uu].v,
            /*c_mod=*/(short)0, acc, /*reuse_a=*/false, /*reuse_b=*/false);
    }
  }

  // C layout: VGPR r, lane -> m = r + 8*half, n = ncol0 + lm
  const float bv = bias[ncol0 + lm];
  if (act == 0) {
    unsigned short* O = (unsigned short*)out;
#pragma unroll
    for (int r = 0; r < 8; ++r) {
      float v = acc[r] + bv;
      v = (v > 0.f) ? v : (__expf(v) - 1.f);                       // ELU
      O[(size_t)(r + 8 * half) * N + ncol0 + lm] = f2bf(v);
    }
  } else {
    float* O = (float*)out;
#pragma unroll
    for (int r = 0; r < 8; ++r) {
      O[(size_t)(r + 8 * half) * N + ncol0 + lm] = sigmoidf_(acc[r] + bv);
    }
  }
}

// ---------------------------------------------------------------------------
// Initial hidden/cell states from noize||category, caps[0]=1
// ---------------------------------------------------------------------------
__global__ __launch_bounds__(256) void k_init(
    const float* __restrict__ noize, const float* __restrict__ category,
    const float* __restrict__ h0W1, const float* __restrict__ h0b1,
    const float* __restrict__ h0W2, const float* __restrict__ h0b2,
    const float* __restrict__ h1W1, const float* __restrict__ h1b1,
    const float* __restrict__ h1W2, const float* __restrict__ h1b2,
    const float* __restrict__ c0W,  const float* __restrict__ c0b,
    const float* __restrict__ c1W,  const float* __restrict__ c1b,
    float* __restrict__ h, float* __restrict__ c, int* __restrict__ caps) {
  __shared__ float sRN[BATCH * 50];
  __shared__ float sHid[BATCH * 200];
  const int tid = threadIdx.x;
  for (int i = tid; i < BATCH * 50; i += 256) {
    int b = i / 50, k = i % 50;
    sRN[i] = (k < 40) ? noize[b * 40 + k] : category[b * 10 + (k - 40)];
  }
  for (int i = tid; i < (LEN_SEQ + 1) * BATCH; i += 256) caps[i] = (i < BATCH) ? 1 : 0;
  __syncthreads();
  for (int L = 0; L < 2; ++L) {
    const float* W1 = L ? h1W1 : h0W1; const float* b1 = L ? h1b1 : h0b1;
    const float* W2 = L ? h1W2 : h0W2; const float* b2 = L ? h1b2 : h0b2;
    const float* cW = L ? c1W : c0W;   const float* cb = L ? c1b : c0b;
    for (int i = tid; i < BATCH * 200; i += 256) {
      int b = i / 200, o = i % 200;
      float s = b1[o];
      for (int k = 0; k < 50; ++k) s += W1[o * 50 + k] * sRN[b * 50 + k];
      sHid[i] = (s > 0.f ? s : 0.f) * INV_BN;                      // ReLU + BN(eval)
    }
    __syncthreads();
    for (int i = tid; i < BATCH * HDN; i += 256) {
      int b = i / HDN, o = i % HDN;
      float s = b2[o];
      for (int k = 0; k < 200; ++k) s += W2[o * 200 + k] * sHid[b * 200 + k];
      h[L * BATCH * HDN + i] = s;
      float sc = cb[o];
      for (int k = 0; k < 50; ++k) sc += cW[o * 50 + k] * sRN[b * 50 + k];
      c[L * BATCH * HDN + i] = sc;
    }
    __syncthreads();
  }
}

// ---------------------------------------------------------------------------
// One sequential step: embed -> MLP -> LSTM0 -> LSTM1 -> log_W1 -> y1 (bf16)
// Single workgroup; all activations staged in LDS.
// ---------------------------------------------------------------------------
__global__ __launch_bounds__(256) void k_step_small(int t,
    const int* __restrict__ caps, const float* __restrict__ emb_matrix,
    const float* __restrict__ embW1, const float* __restrict__ embb1,
    const float* __restrict__ embW2, const float* __restrict__ embb2,
    const float* __restrict__ Wih0, const float* __restrict__ Whh0,
    const float* __restrict__ bih0, const float* __restrict__ bhh0,
    const float* __restrict__ Wih1, const float* __restrict__ Whh1,
    const float* __restrict__ bih1, const float* __restrict__ bhh1,
    const float* __restrict__ logW1, const float* __restrict__ logb1,
    float* __restrict__ h, float* __restrict__ c,
    unsigned short* __restrict__ y1bf) {
  __shared__ float sE1[BATCH * 400];
  __shared__ float sX3[BATCH * 300];   // x300, then h0n @ [0..1600), h1n @ [1600..3200)
  __shared__ float sXe[BATCH * HDN];
  __shared__ float sHP[2 * BATCH * HDN];
  __shared__ int   sTok[BATCH];
  const int tid = threadIdx.x;
  if (tid < BATCH) sTok[tid] = caps[t * BATCH + tid];
  for (int i = tid; i < 2 * BATCH * HDN; i += 256) sHP[i] = h[i];
  __syncthreads();
  for (int i = tid; i < BATCH * 300; i += 256) {
    int b = i / 300, k = i % 300;
    sX3[i] = emb_matrix[(size_t)sTok[b] * 300 + k];
  }
  __syncthreads();
  for (int i = tid; i < BATCH * 400; i += 256) {
    int b = i / 400, o = i % 400;
    float s = embb1[o];
    const float* wr = embW1 + o * 300; const float* xr = sX3 + b * 300;
    for (int k = 0; k < 300; ++k) s += wr[k] * xr[k];
    sE1[i] = s > 0.f ? s : 0.f;
  }
  __syncthreads();
  for (int i = tid; i < BATCH * EMBD; i += 256) {
    int b = i / EMBD, o = i % EMBD;
    float s = embb2[o];
    const float* wr = embW2 + o * 400; const float* xr = sE1 + b * 400;
    for (int k = 0; k < 400; ++k) s += wr[k] * xr[k];
    sXe[i] = s;
  }
  __syncthreads();
  // LSTM cell 0 (gate order i,f,g,o)
  for (int i = tid; i < BATCH * HDN; i += 256) {
    int b = i / HDN, u = i % HDN;
    float g4[4];
    for (int gi = 0; gi < 4; ++gi) {
      int row = gi * HDN + u;
      float s = bih0[row] + bhh0[row];
      const float* wi = Wih0 + row * EMBD; const float* wh = Whh0 + row * HDN;
      const float* xr = sXe + b * EMBD;    const float* hr = sHP + b * HDN;
      for (int k = 0; k < HDN; ++k) s += wi[k] * xr[k] + wh[k] * hr[k];
      g4[gi] = s;
    }
    float c2 = sigmoidf_(g4[1]) * c[i] + sigmoidf_(g4[0]) * tanhf(g4[2]);
    float hn = sigmoidf_(g4[3]) * tanhf(c2);
    c[i] = c2; h[i] = hn; sX3[i] = hn;
  }
  __syncthreads();
  // LSTM cell 1
  for (int i = tid; i < BATCH * HDN; i += 256) {
    int b = i / HDN, u = i % HDN;
    float g4[4];
    for (int gi = 0; gi < 4; ++gi) {
      int row = gi * HDN + u;
      float s = bih1[row] + bhh1[row];
      const float* wi = Wih1 + row * HDN; const float* wh = Whh1 + row * HDN;
      const float* xr = sX3 + b * HDN;    const float* hr = sHP + BATCH * HDN + b * HDN;
      for (int k = 0; k < HDN; ++k) s += wi[k] * xr[k] + wh[k] * hr[k];
      g4[gi] = s;
    }
    float c2 = sigmoidf_(g4[1]) * c[BATCH * HDN + i] + sigmoidf_(g4[0]) * tanhf(g4[2]);
    float hn = sigmoidf_(g4[3]) * tanhf(c2);
    c[BATCH * HDN + i] = c2; h[BATCH * HDN + i] = hn; sX3[BATCH * HDN + i] = hn;
  }
  __syncthreads();
  // y1 = relu(h1n @ log_W1^T + log_b1) -> bf16 padded [16, KPAD2]
  for (int i = tid; i < BATCH * KPAD2; i += 256) {
    int b = i / KPAD2, o = i % KPAD2;
    float s = 0.f;
    if (o < 600) {
      s = logb1[o];
      const float* wr = logW1 + o * HDN; const float* xr = sX3 + BATCH * HDN + b * HDN;
      for (int k = 0; k < HDN; ++k) s += wr[k] * xr[k];
      s = s > 0.f ? s : 0.f;
    }
    y1bf[i] = f2bf(s);
  }
}

// ---------------------------------------------------------------------------
// Post-processing: noise blend, cor mix, dedup x2, argmax -> caps, softmax out
// ---------------------------------------------------------------------------
__global__ __launch_bounds__(1024) void k_post(int t,
    const float* __restrict__ logits, float* __restrict__ probs,
    const float* __restrict__ noise, const float* __restrict__ cor,
    int* __restrict__ caps, float* __restrict__ out) {
  __shared__ float sV[1024];
  __shared__ int   sI[1024];
  __shared__ int   sTok0[BATCH], sTok1[BATCH], sTok2[BATCH];
  __shared__ int   sDamp;
  __shared__ float sRowMax[BATCH], sRowSum[BATCH];
  const int tid = threadIdx.x;
  if (tid < BATCH) {
    sTok0[tid] = caps[t * BATCH + tid];
    int i2 = (t - 1 > 0) ? t - 1 : 0, i3 = (t - 2 > 0) ? t - 2 : 0;
    sTok1[tid] = caps[i2 * BATCH + tid];
    sTok2[tid] = caps[i3 * BATCH + tid];
  }
  __syncthreads();
  const float tf = (float)t;
  const float coef = (t > 3) ? 0.05f : 0.06f;
  const float invden = 1.f / (1.f + tf / 30.f + tf / 20.f + tf / 10.f);
  for (int i = tid; i < BATCH * NTOK; i += 1024) {
    int b = i >> 12, n = i & (NTOK - 1);
    float base = logits[i] * (1.f + coef * tanhf(noise[(size_t)t * BATCH * NTOK + i]));
    float pv = base;
    if (t > 3) {
      float cs = (tf / 10.f) * cor[((size_t)sTok0[b] << 12) + n]
               + (tf / 20.f) * cor[((size_t)sTok1[b] << 12) + n]
               + (tf / 30.f) * cor[((size_t)sTok2[b] << 12) + n];
      pv = (base + logits[n] * cs) * invden;   // logits[n] == l0 broadcast
    }
    probs[i] = pv;
  }
  __syncthreads();
  if (tid == 0) probs[0] = logits[0] * 1.051f;   // probs[0,0]
  __syncthreads();
  // dedup twice: damp batch-0 argmax column if already emitted
  for (int d = 0; d < 2; ++d) {
    float bv = -3.4e38f; int bi = 0;
    for (int n = tid; n < NTOK; n += 1024) {
      float v = probs[n];
      if (v > bv) { bv = v; bi = n; }
    }
    sV[tid] = bv; sI[tid] = bi;
    for (int s = 512; s > 0; s >>= 1) {
      __syncthreads();
      if (tid < s) {
        float ov = sV[tid + s]; int oi = sI[tid + s];
        if (ov > sV[tid] || (ov == sV[tid] && oi < sI[tid])) { sV[tid] = ov; sI[tid] = oi; }
      }
    }
    __syncthreads();
    if (tid == 0) {
      int w00 = sI[0], f = 0;
      if (w00 != 0)
        for (int i = 0; i < (t + 1) * BATCH; ++i)
          if (caps[i] == w00) { f = 1; break; }
      sDamp = f ? w00 : -1;
    }
    __syncthreads();
    if (sDamp >= 0 && tid < BATCH) probs[tid * NTOK + sDamp] *= 0.3f;
    __syncthreads();
  }
  // per-batch argmax -> caps[t+1]; keep row max for softmax
  {
    int row = tid >> 6, l = tid & 63;
    float bv = -3.4e38f; int bi = 0;
    const float* pr = probs + row * NTOK;
    for (int n = l; n < NTOK; n += 64) {
      float v = pr[n];
      if (v > bv) { bv = v; bi = n; }
    }
    sV[tid] = bv; sI[tid] = bi;
    for (int s = 32; s > 0; s >>= 1) {
      __syncthreads();
      if (l < s) {
        float ov = sV[tid + s]; int oi = sI[tid + s];
        if (ov > sV[tid] || (ov == sV[tid] && oi < sI[tid])) { sV[tid] = ov; sI[tid] = oi; }
      }
    }
    __syncthreads();
    if (l == 0) { caps[(t + 1) * BATCH + row] = sI[tid]; sRowMax[row] = sV[tid]; }
  }
  __syncthreads();
  // softmax denominator per row
  {
    int row = tid >> 6, l = tid & 63;
    const float* pr = probs + row * NTOK;
    float acc = 0.f, mx = sRowMax[row];
    for (int n = l; n < NTOK; n += 64) acc += __expf(pr[n] - mx);
    sV[tid] = acc;
    for (int s = 32; s > 0; s >>= 1) {
      __syncthreads();
      if (l < s) sV[tid] += sV[tid + s];
    }
    __syncthreads();
    if (l == 0) sRowSum[row] = sV[tid];
  }
  __syncthreads();
  for (int i = tid; i < BATCH * NTOK; i += 1024) {
    int b = i >> 12, n = i & (NTOK - 1);
    out[(size_t)b * LEN_SEQ * NTOK + (size_t)t * NTOK + n] =
        __expf(probs[i] - sRowMax[b]) / sRowSum[b];
  }
}

// ---------------------------------------------------------------------------
extern "C" void kernel_launch(void* const* d_in, const int* in_sizes, int n_in,
                              void* d_out, int out_size, void* d_ws, size_t ws_size,
                              hipStream_t stream) {
  (void)in_sizes; (void)n_in; (void)out_size; (void)ws_size;
  // setup_inputs() dict order (params flattened in insertion order)
  const float* emb_matrix = (const float*)d_in[0];
  const float* h0W1 = (const float*)d_in[1];  const float* h0b1 = (const float*)d_in[2];
  const float* h0W2 = (const float*)d_in[3];  const float* h0b2 = (const float*)d_in[4];
  const float* h1W1 = (const float*)d_in[5];  const float* h1b1 = (const float*)d_in[6];
  const float* h1W2 = (const float*)d_in[7];  const float* h1b2 = (const float*)d_in[8];
  const float* c0W  = (const float*)d_in[9];  const float* c0b  = (const float*)d_in[10];
  const float* c1W  = (const float*)d_in[11]; const float* c1b  = (const float*)d_in[12];
  const float* embW1 = (const float*)d_in[13]; const float* embb1 = (const float*)d_in[14];
  const float* embW2 = (const float*)d_in[15]; const float* embb2 = (const float*)d_in[16];
  const float* Wih0 = (const float*)d_in[17]; const float* Whh0 = (const float*)d_in[18];
  const float* bih0 = (const float*)d_in[19]; const float* bhh0 = (const float*)d_in[20];
  const float* Wih1 = (const float*)d_in[21]; const float* Whh1 = (const float*)d_in[22];
  const float* bih1 = (const float*)d_in[23]; const float* bhh1 = (const float*)d_in[24];
  const float* logW1 = (const float*)d_in[25]; const float* logb1 = (const float*)d_in[26];
  const float* logW2 = (const float*)d_in[27]; const float* logb2 = (const float*)d_in[28];
  const float* logW3 = (const float*)d_in[29]; const float* logb3 = (const float*)d_in[30];
  const float* cor   = (const float*)d_in[31];
  const float* noize = (const float*)d_in[32];
  const float* category = (const float*)d_in[33];
  const float* noise = (const float*)d_in[34];
  // d_in[35] = truth_caption (unused: dop==0)

  // workspace carve (256B aligned)
  uintptr_t w = (uintptr_t)d_ws;
  auto carve = [&](size_t bytes) -> void* {
    void* p = (void*)w; w += (bytes + 255) & ~(size_t)255; return p;
  };
  unsigned short* W2bf = (unsigned short*)carve((size_t)8192 * KPAD2 * 2);  // ~10.5 MB
  unsigned short* W3bf = (unsigned short*)carve((size_t)4096 * 8192 * 2);   // 67 MB (L2-resident)
  unsigned short* y1bf = (unsigned short*)carve((size_t)BATCH * KPAD2 * 2);
  unsigned short* y2bf = (unsigned short*)carve((size_t)BATCH * 8192 * 2);
  float* logits = (float*)carve((size_t)BATCH * NTOK * 4);
  float* probs  = (float*)carve((size_t)BATCH * NTOK * 4);
  float* hbuf   = (float*)carve((size_t)2 * BATCH * HDN * 4);
  float* cbuf   = (float*)carve((size_t)2 * BATCH * HDN * 4);
  int*   caps   = (int*)carve((size_t)(LEN_SEQ + 1) * BATCH * 4);

  // one-time (per launch) fp32 -> bf16 weight conversion: halves streamed bytes,
  // makes both big matrices fit entirely in the 192 MB L2 across all 20 steps
  {
    long long tot2 = (long long)8192 * KPAD2;
    k_convert_pad<<<dim3((unsigned)((tot2 + 255) / 256)), dim3(256), 0, stream>>>(
        logW2, W2bf, 8192, 600, KPAD2);
    long long tot3 = (long long)4096 * 8192;
    k_convert_pad<<<dim3((unsigned)((tot3 + 255) / 256)), dim3(256), 0, stream>>>(
        logW3, W3bf, 4096, 8192, 8192);
  }
  k_init<<<dim3(1), dim3(256), 0, stream>>>(noize, category,
      h0W1, h0b1, h0W2, h0b2, h1W1, h1b1, h1W2, h1b2,
      c0W, c0b, c1W, c1b, hbuf, cbuf, caps);

  for (int t = 0; t < LEN_SEQ; ++t) {
    k_step_small<<<dim3(1), dim3(256), 0, stream>>>(t, caps, emb_matrix,
        embW1, embb1, embW2, embb2, Wih0, Whh0, bih0, bhh0,
        Wih1, Whh1, bih1, bhh1, logW1, logb1, hbuf, cbuf, y1bf);
    // log_W2: [16,640] x [640 -> 8192], ELU, bf16 out  (512 N-tiles = 512 waves)
    k_wmma_gemm<<<dim3(64), dim3(256), 0, stream>>>(y1bf, W2bf, logb2,
        (void*)y2bf, KPAD2, 8192, 0);
    // log_W3: [16,8192] x [8192 -> 4096], sigmoid, f32 out (256 N-tiles)
    k_wmma_gemm<<<dim3(32), dim3(256), 0, stream>>>(y2bf, W3bf, logb3,
        (void*)logits, 8192, 4096, 1);
    k_post<<<dim3(1), dim3(1024), 0, stream>>>(t, logits, probs, noise, cor,
        caps, (float*)d_out);
  }
}
